// DialogueModel_19284403159368
// MI455X (gfx1250) — matrix operand order repaired and checked by reading, not compiled, not run on hardware
//
#include <hip/hip_runtime.h>

// ===== problem constants (fixed by the reference) =====
#define NN     21845          // nodes
#define DD     384            // feature dim
#define D3     1152           // 3*D
#define EE     174760         // edges
#define NHEAD  6
#define NC1    7
#define NC2    4

typedef __attribute__((ext_vector_type(16))) _Float16 v16h;
typedef __attribute__((ext_vector_type(8)))  _Float16 v8h;
typedef __attribute__((ext_vector_type(8)))  float    v8f;

__device__ __forceinline__ float sigm(float x){ return 1.0f/(1.0f+expf(-x)); }
// monotonic uint key for float atomic-max
__device__ __forceinline__ unsigned fkey(float f){
  unsigned b=__float_as_uint(f); return (b&0x80000000u)? ~b : (b|0x80000000u);
}
__device__ __forceinline__ float funkey(unsigned k){
  unsigned b=(k&0x80000000u)? (k&0x7fffffffu) : ~k; return __uint_as_float(b);
}

// ---------------------------------------------------------------------------
// GEMM  C[M,NCOL] = A[M,384] @ B[384,NCOL]  (f32 storage, f16 WMMA, f32 acc).
// NCOL is a template constant (384 or 1152) so all B loads are immediate-
// offset clauses. tilesN = NCOL/16 is a multiple of 8, so the 8 waves of a
// block always share one 16-row A tile: it is staged once into LDS as f16
// (12 KB) and fragments are read back as two ds_load_b128 per k-step.
// 12 unrolled v_wmma_f32_16x16x32_f16 per 16x16 tile.
// ---------------------------------------------------------------------------
template<int NCOL>
__global__ __launch_bounds__(256) void gemm_wmma(const float* __restrict__ A,
                                                 const float* __restrict__ B,
                                                 float* __restrict__ C, int M)
{
  const int K = 384;
  const int tilesN = NCOL / 16;           // 24 or 72, multiple of 8
  __shared__ __align__(16) _Float16 ldsA[16 * K];   // 12 KB

  int tid  = threadIdx.x;
  int wave = tid >> 5;
  int lane = tid & 31;
  int tile = blockIdx.x * 8 + wave;
  int tm   = tile / tilesN;               // identical for all 8 waves
  int tn   = tile - tm * tilesN;

  // ---- cooperative A-tile stage: 16 rows x 384 cols, f32 -> f16 ----
  {
    int tmb = (blockIdx.x * 8) / tilesN;  // == tm
#pragma unroll
    for (int i = 0; i < 6; ++i) {
      int f4  = tid + 256 * i;            // 0..1535 float4 slots (96 per row)
      int row = f4 / 96;
      int c4  = f4 - row * 96;
      int arow = tmb * 16 + row; if (arow > M - 1) arow = M - 1;  // clamp pad rows
      float4 v = *(const float4*)(A + (size_t)arow * K + c4 * 4);
      _Float16* dp = &ldsA[row * K + c4 * 4];
      dp[0] = (_Float16)v.x; dp[1] = (_Float16)v.y;
      dp[2] = (_Float16)v.z; dp[3] = (_Float16)v.w;
    }
  }
  __syncthreads();

  int hi  = lane >> 4;                    // half-wave select
  int l15 = lane & 15;
  int bcol = tn * 16 + l15;
  v8f acc = {};
#pragma unroll
  for (int k0 = 0; k0 < K; k0 += 32) {
    // A fragment (16x32, contiguous K to match B): lanes 0-15 K=k0..k0+15,
    // lanes 16-31 K=k0+16..k0+31. Two 16B LDS reads.
    int ka = k0 + hi * 16;
    v8h alo = *(const v8h*)&ldsA[l15 * K + ka];
    v8h ahi = *(const v8h*)&ldsA[l15 * K + ka + 8];
    v16h af, bf;
#pragma unroll
    for (int j = 0; j < 8; ++j) { af[j] = alo[j]; af[j + 8] = ahi[j]; }
    // B fragment (32x16): lanes 0-15 K=k0..k0+15, lanes 16-31 K=k0+16..k0+31.
    // NCOL compile-time -> 16 immediate-offset global_load_b32 in one clause.
    int kb = k0 + hi * 16;
    const float* Bp = B + (size_t)kb * NCOL + bcol;
#pragma unroll
    for (int j = 0; j < 16; ++j) bf[j] = (_Float16)Bp[j * NCOL];
    acc = __builtin_amdgcn_wmma_f32_16x16x32_f16(false, af, false, bf,
                                                 (short)0, acc, false, false);
  }
  // C layout: VGPR r -> row r (lanes 0-15) / row r+8 (lanes 16-31)
#pragma unroll
  for (int r = 0; r < 8; ++r) {
    int row = tm * 16 + hi * 8 + r;
    if (row < M) C[(size_t)row * NCOL + bcol] = acc[r];
  }
}

// ---------------------------------------------------------------------------
// TreeLSTM pointwise cell: iou (chunk-local rows) + b_iou -> h,c at row0+lr
// ---------------------------------------------------------------------------
__global__ void lstm_cell(const float* __restrict__ iou, const float* __restrict__ biou,
                          float* __restrict__ h, float* __restrict__ c,
                          int row0, int total, int leaf)
{
  int idx = blockIdx.x * blockDim.x + threadIdx.x;
  if (idx >= total) return;
  int lr = idx / DD, col = idx - lr * DD;
  size_t gr = (size_t)(row0 + lr) * DD + col;
  size_t ib = (size_t)lr * D3 + col;
  float i = iou[ib]        + biou[col];
  float o = iou[ib + DD]   + biou[DD + col];
  float u = iou[ib + 2*DD] + biou[2*DD + col];
  float cpre = leaf ? 0.0f : c[gr];          // c[parent] holds c_agg (reduce4)
  float cn = sigm(i) * tanhf(u) + cpre;
  c[gr] = cn;
  h[gr] = sigm(o) * tanhf(cn);
}

// fc = sigmoid(Uf*ch_h + b) * ch_c    (in-place on the U_f GEMM output)
__global__ void fc_gate(float* __restrict__ fc, const float* __restrict__ ufb,
                        const float* __restrict__ c, int c0, int total)
{
  int idx = blockIdx.x * blockDim.x + threadIdx.x;
  if (idx >= total) return;
  int lr = idx / DD, col = idx - lr * DD;
  float f = sigm(fc[idx] + ufb[col]);
  fc[idx] = f * c[(size_t)(c0 + lr) * DD + col];
}

// BFS tree: children of p are rows 4p+1..4p+4 (contiguous) -> segment sum = reduce4
__global__ void reduce4(const float* __restrict__ h, const float* __restrict__ fc,
                        float* __restrict__ htild, float* __restrict__ c,
                        int p0, int total)
{
  int idx = blockIdx.x * blockDim.x + threadIdx.x;
  if (idx >= total) return;
  int lp = idx / DD, col = idx - lp * DD;
  int p  = p0 + lp;
  size_t hb = (size_t)(4 * p + 1) * DD + col;
  htild[idx] = h[hb] + h[hb + DD] + h[hb + 2*DD] + h[hb + 3*DD];
  size_t fb = (size_t)(4 * lp) * DD + col;
  c[(size_t)p * DD + col] = fc[fb] + fc[fb + DD] + fc[fb + 2*DD] + fc[fb + 3*DD];
}

__global__ void fill_u32(unsigned* __restrict__ p, unsigned v, int n)
{
  int i = blockIdx.x * blockDim.x + threadIdx.x;
  if (i < n) p[i] = v;
}

// e[edge,h] = sum_d attn[h,d] * leakyrelu(ffij + hni[src] + hnj[dst]); atomic segment max
__global__ void edge_e(const float* __restrict__ ffij, const float* __restrict__ hni,
                       const float* __restrict__ hnj, const int* __restrict__ src,
                       const int* __restrict__ dst, const float* __restrict__ attn,
                       float* __restrict__ e_arr, unsigned* __restrict__ emaxkey,
                       int e0, int total)
{
  int idx = blockIdx.x * blockDim.x + threadIdx.x;
  if (idx >= total) return;
  int le = idx / NHEAD, hh = idx - le * NHEAD;
  int e = e0 + le;
  int s = src[e], d = dst[e];
  const float* fp = ffij + (size_t)le * DD + hh * 64;
  const float* np = hni  + (size_t)s  * DD + hh * 64;
  const float* jp = hnj  + (size_t)d  * DD + hh * 64;
  const float* ap = attn + hh * 64;
  float sum = 0.0f;
#pragma unroll 8
  for (int t = 0; t < 64; ++t) {
    float v = fp[t] + np[t] + jp[t];
    v = (v > 0.0f) ? v : 0.2f * v;
    sum += v * ap[t];
  }
  e_arr[(size_t)e * NHEAD + hh] = sum;
  atomicMax(&emaxkey[d * NHEAD + hh], fkey(sum));
}

// decode max keys -> float; empty segments (-inf) -> 0     (in-place reuse)
__global__ void finalize_emax(unsigned* __restrict__ buf, int n)
{
  int i = blockIdx.x * blockDim.x + threadIdx.x;
  if (i >= n) return;
  float m = funkey(buf[i]);
  if (!(m > -3.0e38f)) m = 0.0f;
  buf[i] = __float_as_uint(m);
}

// a = exp(e - emax[dst]); denom[dst] += a  (a overwrites e_arr)
__global__ void edge_a(float* __restrict__ e_arr, const float* __restrict__ demax,
                       float* __restrict__ denom, const int* __restrict__ dst, int total)
{
  int idx = blockIdx.x * blockDim.x + threadIdx.x;
  if (idx >= total) return;
  int e = idx / NHEAD, hh = idx - e * NHEAD;
  int d = dst[e];
  float a = expf(e_arr[idx] - demax[d * NHEAD + hh]);
  e_arr[idx] = a;
  atomicAdd(&denom[d * NHEAD + hh], a);
}

// x[dst, col] += hnode[src, col] * a/denom     (col fastest -> coalesced)
__global__ void edge_agg(const float* __restrict__ a, const float* __restrict__ denom,
                         const float* __restrict__ hnode, const int* __restrict__ src,
                         const int* __restrict__ dst, float* __restrict__ x, int total)
{
  int idx = blockIdx.x * blockDim.x + threadIdx.x;
  if (idx >= total) return;
  int e = idx / DD, col = idx - e * DD;
  int hh = col >> 6;
  int s = src[e], d = dst[e];
  float w = a[(size_t)e * NHEAD + hh] / denom[d * NHEAD + hh];
  atomicAdd(&x[(size_t)d * DD + col], hnode[(size_t)s * DD + col] * w);
}

// Final heads: relu(x+bias) @ lin{1,2} + softmax. One wave per node.
__global__ __launch_bounds__(256) void head_kernel(const float* __restrict__ x,
    const float* __restrict__ bias,
    const float* __restrict__ w1, const float* __restrict__ b1,
    const float* __restrict__ w2, const float* __restrict__ b2,
    float* __restrict__ out1, float* __restrict__ out2, int n)
{
  int wv   = (blockIdx.x * blockDim.x + threadIdx.x) >> 5;
  int lane = threadIdx.x & 31;
  if (wv >= n) return;
  float p1[NC1]; float p2[NC2];
#pragma unroll
  for (int c = 0; c < NC1; ++c) p1[c] = 0.0f;
#pragma unroll
  for (int c = 0; c < NC2; ++c) p2[c] = 0.0f;
  const float* xr = x + (size_t)wv * DD;
#pragma unroll
  for (int j = 0; j < DD / 32; ++j) {
    int col = j * 32 + lane;
    float xv = xr[col] + bias[col];
    xv = (xv > 0.0f) ? xv : 0.0f;
    const float* w1r = w1 + (size_t)col * NC1;
    const float* w2r = w2 + (size_t)col * NC2;
#pragma unroll
    for (int c = 0; c < NC1; ++c) p1[c] += xv * w1r[c];
#pragma unroll
    for (int c = 0; c < NC2; ++c) p2[c] += xv * w2r[c];
  }
#pragma unroll
  for (int off = 16; off > 0; off >>= 1) {
#pragma unroll
    for (int c = 0; c < NC1; ++c) p1[c] += __shfl_down(p1[c], off, 32);
#pragma unroll
    for (int c = 0; c < NC2; ++c) p2[c] += __shfl_down(p2[c], off, 32);
  }
  if (lane == 0) {
    float v1[NC1], m1 = -3.4e38f, s1 = 0.0f;
#pragma unroll
    for (int c = 0; c < NC1; ++c) { v1[c] = p1[c] + b1[c]; m1 = fmaxf(m1, v1[c]); }
#pragma unroll
    for (int c = 0; c < NC1; ++c) { v1[c] = expf(v1[c] - m1); s1 += v1[c]; }
#pragma unroll
    for (int c = 0; c < NC1; ++c) out1[(size_t)wv * NC1 + c] = v1[c] / s1;
    float v2[NC2], m2 = -3.4e38f, s2 = 0.0f;
#pragma unroll
    for (int c = 0; c < NC2; ++c) { v2[c] = p2[c] + b2[c]; m2 = fmaxf(m2, v2[c]); }
#pragma unroll
    for (int c = 0; c < NC2; ++c) { v2[c] = expf(v2[c] - m2); s2 += v2[c]; }
#pragma unroll
    for (int c = 0; c < NC2; ++c) out2[(size_t)wv * NC2 + c] = v2[c] / s2;
  }
}

// ===========================================================================
static inline int cdiv(int a, int b) { return (a + b - 1) / b; }

static void launch_gemm(const float* A, const float* B, float* C, int M, int Nc,
                        hipStream_t s)
{
  int tilesM = (M + 15) / 16;
  if (Nc == D3) {
    int blocks = tilesM * (D3 / 16) / 8;       // tilesN=72 -> exact multiple of 8
    gemm_wmma<D3><<<blocks, 256, 0, s>>>(A, B, C, M);
  } else {
    int blocks = tilesM * (DD / 16) / 8;       // tilesN=24 -> exact multiple of 8
    gemm_wmma<DD><<<blocks, 256, 0, s>>>(A, B, C, M);
  }
}

extern "C" void kernel_launch(void* const* d_in, const int* in_sizes, int n_in,
                              void* d_out, int out_size, void* d_ws, size_t ws_size,
                              hipStream_t stream)
{
  const float* features   = (const float*)d_in[0];
  const float* edge_feats = (const float*)d_in[1];
  const int*   src        = (const int*)d_in[5];
  const int*   dst        = (const int*)d_in[6];
  const float* W_iou      = (const float*)d_in[8];
  const float* U_iou      = (const float*)d_in[9];
  const float* U_f_w      = (const float*)d_in[10];
  const float* U_f_b      = (const float*)d_in[11];
  const float* b_iou      = (const float*)d_in[12];
  const float* fc_ni_w    = (const float*)d_in[13];
  const float* fc_nj_w    = (const float*)d_in[14];
  const float* fc_fij_w   = (const float*)d_in[15];
  const float* fc_node_w  = (const float*)d_in[16];
  const float* attn       = (const float*)d_in[17];
  const float* gat_bias   = (const float*)d_in[18];
  const float* lin1_w     = (const float*)d_in[19];
  const float* lin1_b     = (const float*)d_in[20];
  const float* lin2_w     = (const float*)d_in[21];
  const float* lin2_b     = (const float*)d_in[22];

  // ---- workspace layout (floats), with phase-aliased regions ----
  const size_t ND      = (size_t)NN * DD;                 // 8,388,480
  const size_t OFF_H   = 0;                               // h; later x (agg)
  const size_t OFF_C   = ND;                              // c; later hnode
  const size_t OFF_HNI = 2 * ND;
  const size_t OFF_HNJ = 3 * ND;
  const size_t OFF_BUFA = 4 * ND;                         // iou chunks / ffij chunks
  const size_t SZ_BUFA  = (size_t)4096 * D3;              // 4,718,592
  const size_t OFF_FC   = OFF_BUFA + SZ_BUFA;             // U_f out; later e/emax/denom
  const size_t SZ_FC    = (size_t)16384 * DD;             // 6,291,456
  const size_t OFF_HT   = OFF_FC + SZ_FC;                 // h_tild
  const size_t SZ_HT    = (size_t)4096 * DD;
  const size_t WS_FLOATS = OFF_HT + SZ_HT;                // 46,136,832 (~176 MB)
  if (ws_size < WS_FLOATS * sizeof(float)) return;

  float* ws    = (float*)d_ws;
  float* h_    = ws + OFF_H;
  float* c_    = ws + OFF_C;
  float* hni   = ws + OFF_HNI;
  float* hnj   = ws + OFF_HNJ;
  float* bufA  = ws + OFF_BUFA;
  float* fcbuf = ws + OFF_FC;
  float* htbuf = ws + OFF_HT;

  // ===== Phase 1: TreeLSTM (bottom-up, level synchronous) =====
  // leaves: rows [5461, 21845), iou = features @ W_iou, chunked by 4096 rows
  const int LEAF0 = 5461, LEAFN = 16384, LCH = 4096;
  for (int cb = 0; cb < LEAFN; cb += LCH) {
    launch_gemm(features + (size_t)(LEAF0 + cb) * DD, W_iou, bufA, LCH, D3, stream);
    int tot = LCH * DD;
    lstm_cell<<<cdiv(tot, 256), 256, 0, stream>>>(bufA, b_iou, h_, c_,
                                                  LEAF0 + cb, tot, 1);
  }
  // internal levels t=1..7 (depth d=7-t); nodes at depth d start at (4^d-1)/3
  const int p4[7] = {1, 4, 16, 64, 256, 1024, 4096};
  for (int t = 1; t <= 7; ++t) {
    int d  = 7 - t;
    int p0 = (p4[d] - 1) / 3;
    int pc = p4[d];
    int c0 = 4 * p0 + 1;
    int cc = 4 * pc;
    launch_gemm(h_ + (size_t)c0 * DD, U_f_w, fcbuf, cc, DD, stream);
    fc_gate<<<cdiv(cc * DD, 256), 256, 0, stream>>>(fcbuf, U_f_b, c_, c0, cc * DD);
    reduce4<<<cdiv(pc * DD, 256), 256, 0, stream>>>(h_, fcbuf, htbuf, c_, p0, pc * DD);
    launch_gemm(htbuf, U_iou, bufA, pc, D3, stream);
    lstm_cell<<<cdiv(pc * DD, 256), 256, 0, stream>>>(bufA, b_iou, h_, c_,
                                                      p0, pc * DD, 0);
  }

  // ===== Phase 2: EdgeGAT =====
  float* hnode = c_;                       // c dead after TreeLSTM
  launch_gemm(h_, fc_ni_w,   hni,   NN, DD, stream);
  launch_gemm(h_, fc_nj_w,   hnj,   NN, DD, stream);
  launch_gemm(h_, fc_node_w, hnode, NN, DD, stream);

  float*    e_arr   = fcbuf;                                    // E*6
  unsigned* emaxkey = (unsigned*)(fcbuf + (size_t)EE * NHEAD);  // N*6
  float*    denom   = fcbuf + (size_t)EE * NHEAD + (size_t)NN * NHEAD;
  float*    x_      = h_;                  // h dead after the three projections

  fill_u32<<<cdiv(NN * DD, 256), 256, 0, stream>>>((unsigned*)x_, 0u, NN * DD);
  fill_u32<<<cdiv(NN * NHEAD, 256), 256, 0, stream>>>(emaxkey, 0x007FFFFFu, NN * NHEAD);
  fill_u32<<<cdiv(NN * NHEAD, 256), 256, 0, stream>>>((unsigned*)denom, 0u, NN * NHEAD);

  const int ECH = 8192;                    // edge chunk (fits bufA: 8192*384)
  for (int cs = 0; cs < EE; cs += ECH) {
    int cc = (EE - cs < ECH) ? (EE - cs) : ECH;
    launch_gemm(edge_feats + (size_t)cs * DD, fc_fij_w, bufA, cc, DD, stream);
    edge_e<<<cdiv(cc * NHEAD, 256), 256, 0, stream>>>(bufA, hni, hnj, src, dst,
                                                      attn, e_arr, emaxkey,
                                                      cs, cc * NHEAD);
  }
  finalize_emax<<<cdiv(NN * NHEAD, 256), 256, 0, stream>>>(emaxkey, NN * NHEAD);
  edge_a<<<cdiv(EE * NHEAD, 256), 256, 0, stream>>>(e_arr, (const float*)emaxkey,
                                                    denom, dst, EE * NHEAD);
  edge_agg<<<cdiv(EE * DD, 256), 256, 0, stream>>>(e_arr, denom, hnode, src, dst,
                                                   x_, EE * DD);

  // ===== Phase 3: heads =====
  float* out1 = (float*)d_out;
  float* out2 = out1 + (size_t)NN * NC1;
  head_kernel<<<cdiv(NN * 32, 256), 256, 0, stream>>>(x_, gat_bias, lin1_w, lin1_b,
                                                      lin2_w, lin2_b, out1, out2, NN);
}
